// SimpleEdgeBlock_79551384256581
// MI455X (gfx1250) — matrix operand order repaired and checked by reading, not compiled, not run on hardware
//
#include <hip/hip_runtime.h>

typedef __attribute__((ext_vector_type(16))) _Float16 v16h;
typedef __attribute__((ext_vector_type(8)))  float    v8f;

#define EPB     16      // edges per block (WMMA M tile)
#define THREADS 128     // 4 waves (wave32)
#define XROW    288     // flat irrep row length
#define BROW    1040    // padded LDS row stride (halves) for B tile
#define NSLICE  35
#define WTOT    (11 * 32 * 1024)   // f16 transposed weights in ws
#define CGTOT   363                // dense per-slice CG matrices in ws

// ---- real-basis Wigner-3j nonzeros, ordered by (l3,k)-slice ---------------
struct Nz { unsigned char i, j; float v; };

#define S3  0.57735027f   // 1/sqrt(3)
#define S5  0.44721360f   // 1/sqrt(5)
#define R10 0.31622776f   // 1/sqrt(10)
#define C2  0.36514837f   // 2/sqrt(30)
#define C1  0.18257419f   // 1/sqrt(30)
#define T22 0.23904572f
#define H22 0.11952286f
#define U22 0.20702007f

__constant__ Nz SLNZ[83] = {
  /* s0  (p0 ,k0)*/ {0,0,1.0f},
  /* s1  (p4 ,k0)*/ {0,0,S3},{1,1,S3},{2,2,S3},
  /* s2  (p9 ,k0)*/ {0,0,S5},{1,1,S5},{2,2,S5},{3,3,S5},{4,4,S5},
  /* s3  (p1 ,k0)*/ {0,0,S3},
  /* s4  (p3 ,k0)*/ {0,0,S3},
  /* s5  (p6 ,k0)*/ {2,0,R10},{1,1,R10},{0,4,-R10},{0,2,-C1},
  /* s6  (p8 ,k0)*/ {0,2,R10},{1,1,R10},{4,0,-R10},{2,0,-C1},
  /* s7  (p1 ,k1)*/ {0,1,S3},
  /* s8  (p3 ,k1)*/ {1,0,S3},
  /* s9  (p6 ,k1)*/ {0,1,R10},{2,3,R10},{1,2,C2},
  /* s10 (p8 ,k1)*/ {1,0,R10},{3,2,R10},{2,1,C2},
  /* s11 (p1 ,k2)*/ {0,2,S3},
  /* s12 (p3 ,k2)*/ {2,0,S3},
  /* s13 (p6 ,k2)*/ {0,0,R10},{1,3,R10},{2,4,R10},{2,2,-C1},
  /* s14 (p8 ,k2)*/ {0,0,R10},{3,1,R10},{4,2,R10},{2,2,-C1},
  /* s15 (p2 ,k0)*/ {0,0,S5},
  /* s16 (p5 ,k0)*/ {2,0,R10},{0,2,R10},
  /* s17 (p7 ,k0)*/ {0,0,S5},
  /* s18 (p10,k0)*/ {2,0,-T22},{0,2,-T22},{1,3,U22},{3,1,U22},
  /* s19 (p2 ,k1)*/ {0,1,S5},
  /* s20 (p5 ,k1)*/ {0,1,R10},{1,0,R10},
  /* s21 (p7 ,k1)*/ {1,0,S5},
  /* s22 (p10,k1)*/ {2,1,H22},{1,2,H22},{4,1,-U22},{1,4,-U22},{0,3,U22},{3,0,U22},
  /* s23 (p2 ,k2)*/ {0,2,S5},
  /* s24 (p5 ,k2)*/ {1,1,C2},{2,2,-C1},{0,0,-C1},
  /* s25 (p7 ,k2)*/ {2,0,S5},
  /* s26 (p10,k2)*/ {2,2,T22},{0,0,-T22},{1,1,H22},{3,3,H22},{4,4,-T22},
  /* s27 (p2 ,k3)*/ {0,3,S5},
  /* s28 (p5 ,k3)*/ {2,1,R10},{1,2,R10},
  /* s29 (p7 ,k3)*/ {3,0,S5},
  /* s30 (p10,k3)*/ {2,3,H22},{3,2,H22},{4,3,U22},{3,4,U22},{0,1,U22},{1,0,U22},
  /* s31 (p2 ,k4)*/ {0,4,S5},
  /* s32 (p5 ,k4)*/ {2,2,R10},{0,0,-R10},
  /* s33 (p7 ,k4)*/ {4,0,S5},
  /* s34 (p10,k4)*/ {2,4,-T22},{4,2,-T22},{1,1,-U22},{3,3,U22}
};
__constant__ short SL_OFF[36] = {0,1,4,9,10,11,15,19,20,21,24,27,28,29,33,37,
                                 38,40,41,45,46,48,49,55,56,59,60,65,66,68,69,
                                 75,76,78,79,83};
// path id of each slice (slice order = group-major: (l3,k) then feeding paths)
__constant__ unsigned char SL_P[NSLICE] = {
  0,4,9, 1,3,6,8, 1,3,6,8, 1,3,6,8,
  2,5,7,10, 2,5,7,10, 2,5,7,10, 2,5,7,10, 2,5,7,10};
// dense CG matrix offset per slice (D2 x D1, row-major), total 363 floats
__constant__ short CG_OFF[NSLICE] = {
  0,1,10, 35,38,41,56, 71,74,77,92, 107,110,113,128,
  143,148,157,162, 187,192,201,206, 231,236,245,250,
  275,280,289,294, 319,324,333,338};

__constant__ unsigned char P_L1[11] = {0,0,0,1,1,1,1,2,2,2,2};
__constant__ unsigned char P_L2[11] = {0,1,2,0,1,1,2,0,1,2,2};
__constant__ int           DIMS3[3] = {1,3,5};
__constant__ int           OFFS3[3] = {0,32,128};
// alpha(l3) = sqrt((2*l3+1) / (NPATH[l3]*32*32))
__constant__ float        ALPHA3[3] = {0.018042196f, 0.027063294f, 0.034938562f};

// 9 output groups (l3,k); each accumulates its feeding paths in the WMMA C regs
__constant__ unsigned char G_L3[9]  = {0,1,1,1,2,2,2,2,2};
__constant__ unsigned char G_K[9]   = {0,0,1,2,0,1,2,3,4};
__constant__ unsigned char G_NP[9]  = {3,4,4,4,4,4,4,4,4};

// ---- pre-kernel: f16 W[p][u][v][w] -> Wt[p][w][uv], plus dense CG build ---
__global__ __launch_bounds__(256)
void tp_prepack(const float* __restrict__ w, _Float16* __restrict__ wt,
                float* __restrict__ cg) {
  int idx = blockIdx.x * 256 + threadIdx.x;
  if (idx < WTOT) {
    int p  = idx >> 15;
    int wc = (idx >> 10) & 31;
    int u  = (idx >> 5)  & 31;
    int v  = idx & 31;
    wt[idx] = (_Float16)w[(((p * 32 + u) * 32 + v) * 32) + wc];
  }
  if (blockIdx.x == 0) {
    for (int i = threadIdx.x; i < CGTOT; i += 256) cg[i] = 0.f;
    __syncthreads();
    if (threadIdx.x == 0) {
      for (int s = 0; s < NSLICE; ++s) {
        int d1 = DIMS3[P_L1[SL_P[s]]];
        for (int n = SL_OFF[s]; n < SL_OFF[s + 1]; ++n) {
          Nz z = SLNZ[n];
          cg[CG_OFF[s] + (int)z.j * d1 + (int)z.i] = z.v;
        }
      }
    }
  }
}

// ---- B-phase: fully unrolled per (d1,d2); v-outer so x2 is loaded once ----
template <int D1, int D2>
__device__ __forceinline__ void bphase_t(const float* __restrict__ cg,
                                         const float* __restrict__ x1r,
                                         const float* __restrict__ x2r,
                                         _Float16* __restrict__ bout, int ub) {
  float cgr[D2][D1];
#pragma unroll
  for (int j = 0; j < D2; ++j)
#pragma unroll
    for (int i = 0; i < D1; ++i) cgr[j][i] = cg[j * D1 + i];

  float s[4][D2];
#pragma unroll
  for (int uu = 0; uu < 4; ++uu) {
    const float* x1u = x1r + (ub + uu) * D1;
#pragma unroll
    for (int j = 0; j < D2; ++j) {
      float acc = 0.f;
#pragma unroll
      for (int i = 0; i < D1; ++i) acc += cgr[j][i] * x1u[i];
      s[uu][j] = acc;
    }
  }
#pragma unroll 4
  for (int v = 0; v < 32; v += 2) {
    float a0[D2], a1[D2];
#pragma unroll
    for (int j = 0; j < D2; ++j) {
      a0[j] = x2r[v * D2 + j];
      a1[j] = x2r[(v + 1) * D2 + j];
    }
#pragma unroll
    for (int uu = 0; uu < 4; ++uu) {
      float p0 = 0.f, p1 = 0.f;
#pragma unroll
      for (int j = 0; j < D2; ++j) {
        p0 += s[uu][j] * a0[j];
        p1 += s[uu][j] * a1[j];
      }
      union { _Float16 h[2]; unsigned int u; } pk;
      pk.h[0] = (_Float16)p0;
      pk.h[1] = (_Float16)p1;
      *(unsigned int*)(bout + (uu << 5) + v) = pk.u;  // packed b32 store
    }
  }
}

__device__ __forceinline__ void bphase(int d1, int d2, const float* cg,
                                       const float* x1r, const float* x2r,
                                       _Float16* bout, int ub) {
  switch (d1 * 8 + d2) {
    case  9: bphase_t<1, 1>(cg, x1r, x2r, bout, ub); break;
    case 11: bphase_t<1, 3>(cg, x1r, x2r, bout, ub); break;
    case 13: bphase_t<1, 5>(cg, x1r, x2r, bout, ub); break;
    case 25: bphase_t<3, 1>(cg, x1r, x2r, bout, ub); break;
    case 27: bphase_t<3, 3>(cg, x1r, x2r, bout, ub); break;
    case 29: bphase_t<3, 5>(cg, x1r, x2r, bout, ub); break;
    case 41: bphase_t<5, 1>(cg, x1r, x2r, bout, ub); break;
    case 43: bphase_t<5, 3>(cg, x1r, x2r, bout, ub); break;
    default: bphase_t<5, 5>(cg, x1r, x2r, bout, ub); break;
  }
}

// ---- main kernel: 16 edges / block, 4 waves, double-buffered B tile -------
__global__ __launch_bounds__(THREADS)
void tp_fctp_kernel(const float* __restrict__ x1g, const float* __restrict__ x2g,
                    const _Float16* __restrict__ Wt, const float* __restrict__ CGf,
                    float* __restrict__ outg, int E) {
  extern __shared__ char smem_raw[];
  float*    X1  = (float*)smem_raw;              // 16*288 f32
  float*    X2  = X1 + EPB * XROW;               // 16*288 f32
  float*    OUT = X2 + EPB * XROW;               // 16*288 f32 accumulators
  _Float16* Bb0 = (_Float16*)(OUT + EPB * XROW); // 16*1040 f16 (buffer 0)
  _Float16* Bb1 = Bb0 + EPB * BROW;              // 16*1040 f16 (buffer 1)

  const int tid = threadIdx.x;
  const int e0  = blockIdx.x * EPB;

  // stage inputs (coalesced), zero output accumulators
  for (int idx = tid; idx < EPB * XROW; idx += THREADS) {
    int ge = e0 + idx / XROW;
    float a = 0.f, b = 0.f;
    if (ge < E) {
      a = x1g[(size_t)e0 * XROW + idx];
      b = x2g[(size_t)e0 * XROW + idx];
    }
    X1[idx] = a; X2[idx] = b; OUT[idx] = 0.f;
  }
  __syncthreads();

  const int lane  = tid & 31;
  const int wid   = tid >> 5;
  const int nhalf = wid & 1;        // which 16 of the 32 w-channels
  const int khalf = wid >> 1;       // which 512 of K=1024
  const int row   = lane & 15;
  const int hi    = lane >> 4;
  const int eB    = tid >> 3;       // edge handled in B-phase
  const int ub    = (tid & 7) << 2; // 4-u group in B-phase

  // prologue: produce slice 0 into buffer 0
  {
    const int p = SL_P[0];
    bphase(DIMS3[P_L1[p]], DIMS3[P_L2[p]], CGf + CG_OFF[0],
           X1 + eB * XROW + OFFS3[P_L1[p]], X2 + eB * XROW + OFFS3[P_L2[p]],
           Bb0 + eB * BROW + (ub << 5), ub);
  }
  __syncthreads();

  int s = 0;
  for (int g = 0; g < 9; ++g) {
    const int l3 = G_L3[g];
    const int np = G_NP[g];
    v8f c = {0.f, 0.f, 0.f, 0.f, 0.f, 0.f, 0.f, 0.f};

    for (int pi = 0; pi < np; ++pi, ++s) {
      const _Float16* cur = (s & 1) ? Bb1 : Bb0;
      // ---- issue WMMA chain for slice s (XDL co-executes with VALU below)
      {
        const int p = SL_P[s];
        const _Float16* arow = cur + row * BROW;
        const _Float16* wrow = Wt + ((size_t)p << 15)
                                  + (size_t)((nhalf << 4) + row) * 1024;
        const int tbase = khalf << 4;
#pragma unroll
        for (int t = 0; t < 16; ++t) {
          const int koff = (tbase + t) << 5;
          union { v16h h; uint4 q[2]; } a;
          a.q[0] = *(const uint4*)(arow + koff + (hi << 4));
          a.q[1] = *(const uint4*)(arow + koff + (hi << 4) + 8);
          v16h bf = *(const v16h*)(wrow + koff + (hi << 4));
          c = __builtin_amdgcn_wmma_f32_16x16x32_f16(
                false, a.h, false, bf, (short)0, c, false, false);
        }
      }
      // ---- produce next slice's B tile into the other buffer (VALU)
      if (s + 1 < NSLICE) {
        const int pn = SL_P[s + 1];
        _Float16* nxt = (s & 1) ? Bb0 : Bb1;
        bphase(DIMS3[P_L1[pn]], DIMS3[P_L2[pn]], CGf + CG_OFF[s + 1],
               X1 + eB * XROW + OFFS3[P_L1[pn]],
               X2 + eB * XROW + OFFS3[P_L2[pn]],
               nxt + eB * BROW + (ub << 5), ub);
      }
      __syncthreads();
    }

    // writeout: merge the two K-half partials via LDS float atomics, scaled
    const float al   = ALPHA3[l3];
    const int   d3   = DIMS3[l3];
    const int   wcol = OFFS3[l3] + ((nhalf << 4) + row) * d3 + (int)G_K[g];
#pragma unroll
    for (int r = 0; r < 8; ++r) {
      atomicAdd(&OUT[(r + (hi << 3)) * XROW + wcol], al * c[r]);
    }
  }

  __syncthreads();
  for (int idx = tid; idx < EPB * XROW; idx += THREADS) {
    int ge = e0 + idx / XROW;
    if (ge < E) outg[(size_t)e0 * XROW + idx] = OUT[idx];
  }
}

extern "C" void kernel_launch(void* const* d_in, const int* in_sizes, int n_in,
                              void* d_out, int out_size, void* d_ws, size_t ws_size,
                              hipStream_t stream) {
  (void)n_in; (void)out_size; (void)ws_size;
  const float* x1 = (const float*)d_in[0];
  const float* x2 = (const float*)d_in[1];
  const float* w  = (const float*)d_in[2];
  float* out = (float*)d_out;

  const int E = in_sizes[0] / XROW;

  // workspace: [0, 704KB) f16 transposed weights; then 363 f32 dense CG mats
  _Float16* Wt  = (_Float16*)d_ws;
  float*    CGf = (float*)((char*)d_ws + (size_t)WTOT * sizeof(_Float16));
  tp_prepack<<<(WTOT + 255) / 256, 256, 0, stream>>>(w, Wt, CGf);

  const int nblocks = (E + EPB - 1) / EPB;
  const size_t smem = (size_t)(3 * EPB * XROW) * sizeof(float)
                    + (size_t)(2 * EPB * BROW) * sizeof(_Float16); // 121856 B
  tp_fctp_kernel<<<nblocks, THREADS, smem, stream>>>(x1, x2, Wt, CGf, out, E);
}
